// S4D_28570122453220
// MI455X (gfx1250) — matrix operand order repaired and checked by reading, not compiled, not run on hardware
//
#include <hip/hip_runtime.h>

// ---------------- problem constants ----------------
#define HH 512
#define N2 32
#define LL 4096
#define BB 8
#define Q  32            // chunk length (matches WMMA K=32 for f16)
#define NC (LL / Q)      // 128 chunks per row

typedef _Float16 f16;
typedef _Float16 v16h  __attribute__((ext_vector_type(16)));
typedef _Float16 f16x8 __attribute__((ext_vector_type(8)));
typedef _Float16 f16x2 __attribute__((ext_vector_type(2)));
typedef float    v8f   __attribute__((ext_vector_type(8)));

// CDNA5 async global->LDS staging (ASYNCcnt path, ISA 15.18.3 op 98).
#define USE_ASYNC_STAGING 1

#if USE_ASYNC_STAGING
__device__ __forceinline__ void async_g2l_b128(unsigned lds_off, const void* gptr) {
  asm volatile("global_load_async_to_lds_b128 %0, %1, off"
               :: "v"(lds_off), "v"((unsigned long long)(uintptr_t)gptr)
               : "memory");
}
__device__ __forceinline__ void wait_asynccnt0() {
  asm volatile("s_wait_asynccnt 0x0" ::: "memory");
}
#endif

// ---------------- per-head workspace layout (bytes) ----------------
#define OFF_T0 0          // 32x32 f16 lower-Toeplitz (causal intra)      2048 B
#define OFF_T1 2048       // 32x32 f16 upper-Toeplitz (anticausal intra)  2048 B
#define OFF_WF 4096       // 64x32 f16 forward injection                  4096 B
#define OFF_WB 8192       // 64x32 f16 backward injection                 4096 B
#define OFF_OF 12288      // 32x64 f16 forward state readout              4096 B
#define OFF_OB 16384      // 32x64 f16 backward state readout             4096 B
#define OFF_WQ 20480      // 32 re + 32 im f32 of w^Q                      256 B
#define H_STRIDE 20736    // per-head stride (512 heads => ~10.1 MB, L2-resident)

// Dynamic LDS (bytes):
//  [0,20480)      : T0,T1,WF,WB,OF,OB staged params (same order as ws)
//  [20480,20736)  : wQ (64 f32)
//  [20736,37120)  : u row f32 (4096)
//  [37120,53504)  : y row f32 (4096)
//  [53504,61696)  : Uc  f16  B-swizzled [2 rowgroups][NC][16]
//  [61696,78080)  : Sf  f16  B-swizzled [4 rowgroups][NC][16]
//  [78080,94464)  : Sb  f16  B-swizzled [4 rowgroups][NC][16]
#define SMEM_MAIN 94464

// B-swizzled addressing: element (row K, col) lives at ((K>>4)*NC + col)*16 + (K&15).
// => a WMMA B fragment (lane g,m; rows brg+g, col bcol+m) is 16 contiguous halves.

// =====================================================================
// Kernel 1: per-head parameter precompute.  One wave per head; lane = pole n.
// =====================================================================
__global__ __launch_bounds__(32) void s4d_precompute(
    const float* __restrict__ log_dt,
    const float* __restrict__ log_neg_A_re,
    const float* __restrict__ A_im,
    const float* __restrict__ Bp,    // [H,N2,2]
    const float* __restrict__ Cp,    // [2,H,N2,2]
    char* __restrict__ ws)
{
  __shared__ float pr[Q + 1][N2], pi[Q + 1][N2];   // w^j powers, j=0..Q
  __shared__ float c0r[N2], c0i[N2], c1r[N2], c1i[N2];
  __shared__ float k0[Q], k1[Q];

  const int h = blockIdx.x;
  const int n = threadIdx.x;

  const float dt = __expf(log_dt[h]);
  const float Ar = -__expf(log_neg_A_re[h * N2 + n]);
  const float Ai = A_im[h * N2 + n];
  const float dr = dt * Ar, di = dt * Ai;
  const float er = __expf(dr);
  const float wr = er * __cosf(di);
  const float wi = er * __sinf(di);

  // g = (exp(dtA) - 1) / A
  const float den  = Ar * Ar + Ai * Ai;
  const float numr = wr - 1.0f, numi = wi;
  const float gr = (numr * Ar + numi * Ai) / den;
  const float gi = (numi * Ar - numr * Ai) / den;

  const float Br = Bp[(h * N2 + n) * 2 + 0];
  const float Bi = Bp[(h * N2 + n) * 2 + 1];

  // Cc[c] = (C[c]*B) * g
  {
    const float C0r = Cp[((0 * HH + h) * N2 + n) * 2 + 0];
    const float C0i = Cp[((0 * HH + h) * N2 + n) * 2 + 1];
    const float C1r = Cp[((1 * HH + h) * N2 + n) * 2 + 0];
    const float C1i = Cp[((1 * HH + h) * N2 + n) * 2 + 1];
    float t0r = C0r * Br - C0i * Bi, t0i = C0r * Bi + C0i * Br;
    float t1r = C1r * Br - C1i * Bi, t1i = C1r * Bi + C1i * Br;
    c0r[n] = t0r * gr - t0i * gi;  c0i[n] = t0r * gi + t0i * gr;
    c1r[n] = t1r * gr - t1i * gi;  c1i[n] = t1r * gi + t1i * gr;
  }

  // powers w^0..w^Q
  {
    float ar = 1.0f, ai = 0.0f;
    for (int j = 0; j <= Q; ++j) {
      pr[j][n] = ar; pi[j][n] = ai;
      float t = ar * wr - ai * wi;
      ai = ar * wi + ai * wr;
      ar = t;
    }
  }
  __syncthreads();

  // Toeplitz coefficients k0[p], k1[p] = 2*Re(sum_n Cc[n] * w[n]^p); lane = p
  {
    float a0 = 0.f, a1 = 0.f;
    for (int m = 0; m < N2; ++m) {
      a0 += c0r[m] * pr[n][m] - c0i[m] * pi[n][m];
      a1 += c1r[m] * pr[n][m] - c1i[m] * pi[n][m];
    }
    k0[n] = 2.0f * a0;
    k1[n] = 2.0f * a1;
  }
  __syncthreads();

  char* hp = ws + (size_t)h * H_STRIDE;
  f16* T0 = (f16*)(hp + OFF_T0);
  f16* T1 = (f16*)(hp + OFF_T1);
  f16* WF = (f16*)(hp + OFF_WF);
  f16* WB = (f16*)(hp + OFF_WB);
  f16* OF = (f16*)(hp + OFF_OF);
  f16* OB = (f16*)(hp + OFF_OB);
  float* WQ = (float*)(hp + OFF_WQ);

  // lane = row i of the Toeplitz matrices
  for (int j = 0; j < Q; ++j) {
    T0[n * Q + j] = (f16)((j <= n) ? k0[n - j] : 0.0f);
    T1[n * Q + j] = (f16)((j > n)  ? k1[j - n - 1] : 0.0f);
  }
  // lane = pole n : injection rows (2n = Re, 2n+1 = Im)
  for (int j = 0; j < Q; ++j) {
    WF[(2 * n    ) * Q + j] = (f16)pr[Q - 1 - j][n];
    WF[(2 * n + 1) * Q + j] = (f16)pi[Q - 1 - j][n];
    WB[(2 * n    ) * Q + j] = (f16)pr[j][n];
    WB[(2 * n + 1) * Q + j] = (f16)pi[j][n];
  }
  // readout: y_inter_f[i] = 2Re(Cc0 * w^{i+1} * S),  y_inter_b[i] = 2Re(Cc1 * w^{Q-1-i} * T)
  for (int i = 0; i < Q; ++i) {
    float fr = pr[i + 1][n], fi = pi[i + 1][n];
    OF[i * 64 + 2 * n    ] = (f16)( 2.0f * (c0r[n] * fr - c0i[n] * fi));
    OF[i * 64 + 2 * n + 1] = (f16)(-2.0f * (c0r[n] * fi + c0i[n] * fr));
    float qr = pr[Q - 1 - i][n], qi = pi[Q - 1 - i][n];
    OB[i * 64 + 2 * n    ] = (f16)( 2.0f * (c1r[n] * qr - c1i[n] * qi));
    OB[i * 64 + 2 * n + 1] = (f16)(-2.0f * (c1r[n] * qi + c1i[n] * qr));
  }
  WQ[n]      = pr[Q][n];
  WQ[N2 + n] = pi[Q][n];
}

// =====================================================================
// WMMA helper: one 16x16x32 f16 MACC.
//   A (row-major in LDS): per-lane K pattern {16g..16g+15} -> 2x b128.
//   B (B-swizzled in LDS): per-lane 16 contiguous halves -> 2x b128.
// =====================================================================
__device__ __forceinline__ v8f wmma_acc(const f16* __restrict__ A, int lda, int arow, int acol,
                                        const f16* __restrict__ Bsw, int brg, int bcol,
                                        v8f acc, int g, int m)
{
  v16h a;
#pragma unroll
  for (int e = 0; e < 16; ++e) {
    const int v = e >> 1, hb = e & 1;
    const int k = 16 * g + 2 * v + hb;
    a[e] = A[(arow + m) * lda + acol + k];
  }
  const v16h bf = *(const v16h*)&Bsw[((brg + g) * NC + bcol + m) * 16];
  return __builtin_amdgcn_wmma_f32_16x16x32_f16(false, a, false, bf, (short)0, acc, false, false);
}

// =====================================================================
// Kernel 2: one workgroup per (batch, head) row.  4 waves (wave32).
// =====================================================================
__global__ __launch_bounds__(128) void s4d_main(
    const float* __restrict__ u,
    const float* __restrict__ Dv,
    const char*  __restrict__ ws,
    float* __restrict__ y)
{
  extern __shared__ char smem[];
  f16*   pT0 = (f16*)smem;            // 1024 halves
  f16*   pT1 = pT0 + 1024;
  f16*   pWF = pT1 + 1024;            // 2048 halves
  f16*   pWB = pWF + 2048;
  f16*   pOF = pWB + 2048;            // 2048 halves
  f16*   pOB = pOF + 2048;
  float* pWQ = (float*)(pOB + 2048);  // 64 f32
  float* uS  = pWQ + 64;              // 4096 f32
  float* yS  = uS + 4096;             // 4096 f32
  f16*   Uc  = (f16*)(yS + 4096);     // B-swizzled [2][NC][16]
  f16*   Sf  = Uc + 2 * NC * 16;      // B-swizzled [4][NC][16]
  f16*   Sb  = Sf + 4 * NC * 16;      // B-swizzled [4][NC][16]

  const int tid  = threadIdx.x;
  const int wave = tid >> 5;
  const int lane = tid & 31;
  const int g    = lane >> 4;
  const int m    = lane & 15;
  const int h    = blockIdx.x >> 3;   // 8 consecutive blocks share head params (L2-hot)
  const int b    = blockIdx.x & 7;

  const char* hp = ws + (size_t)h * H_STRIDE;
  const float* urow = u + ((size_t)b * HH + h) * LL;

  // ---- stage per-head params + u row into LDS ----
#if USE_ASYNC_STAGING
  {
    // ASYNCcnt-tracked DMA: memory -> LDS without a VGPR round-trip.
    const unsigned lds_par = (unsigned)(uintptr_t)smem;
    for (int i = tid; i < 1280; i += 128)                 // 20480 B of params
      async_g2l_b128(lds_par + i * 16, hp + (size_t)i * 16);
    const unsigned lds_u = (unsigned)(uintptr_t)uS;
    for (int i = tid; i < LL / 4; i += 128)               // 16384 B of u
      async_g2l_b128(lds_u + i * 16, urow + i * 4);
    const float* wq = (const float*)(hp + OFF_WQ);
    if (tid < 64) pWQ[tid] = wq[tid];
    wait_asynccnt0();                                     // s_wait_asynccnt 0
  }
#else
  {
    const uint4* src = (const uint4*)hp;
    uint4* dst = (uint4*)smem;
    for (int i = tid; i < 1280; i += 128) dst[i] = src[i];
    const float* wq = (const float*)(hp + OFF_WQ);
    if (tid < 64) pWQ[tid] = wq[tid];
    const float4* us = (const float4*)urow;
    float4* ud = (float4*)uS;
    for (int i = tid; i < LL / 4; i += 128) ud[i] = us[i];
  }
#endif
  __syncthreads();

  // ---- build B-swizzled f16 input: element (row j, col c) = u[c*Q + j] ----
  for (int i = tid; i < 2 * NC; i += 128) {
    const int c  = i >> 1;
    const int rg = i & 1;
    const float* src = uS + c * Q + rg * 16;
    f16x8 lo, hi;
#pragma unroll
    for (int e = 0; e < 8; ++e) { lo[e] = (f16)src[e]; hi[e] = (f16)src[8 + e]; }
    f16x8* dst = (f16x8*)&Uc[(rg * NC + c) * 16];
    dst[0] = lo; dst[1] = hi;
  }
  __syncthreads();

  // ---- GEMM1: injections  Inj(64 x NC) = Winj(64x32) @ Uc(32xNC), both dirs ----
  for (int t = wave; t < 64; t += 4) {
    const int dir = t >> 5;
    const int mt  = (t >> 3) & 3;
    const int nt  = t & 7;
    const f16* W  = dir ? pWB : pWF;
    v8f acc = {};
    acc = wmma_acc(W, Q, 16 * mt, 0, Uc, 0, 16 * nt, acc, g, m);
    // state row k = 16*mt + 8*g + r  ->  rowgroup mt, e = 8*g + r : one b128 store
    f16x8 pk;
#pragma unroll
    for (int r = 0; r < 8; ++r) pk[r] = (f16)acc[r];
    f16* S = dir ? Sb : Sf;
    *(f16x8*)&S[(mt * NC + 16 * nt + m) * 16 + 8 * g] = pk;
  }
  __syncthreads();

  // ---- chunk-level scans (f32 states in registers, in-place over Inj) ----
  if (wave == 0) {                       // forward: S_in[c], c = 0..NC-1
    const float wr = pWQ[lane], wi = pWQ[N2 + lane];
    float sr = 0.f, si = 0.f;
    f16* base = Sf + ((lane >> 3) * NC) * 16 + ((2 * lane) & 15);
    for (int c = 0; c < NC; ++c) {
      const f16x2 inj = *(const f16x2*)&base[c * 16];
      f16x2 out; out[0] = (f16)sr; out[1] = (f16)si;
      *(f16x2*)&base[c * 16] = out;
      const float t = wr * sr - wi * si + (float)inj[0];
      si = wi * sr + wr * si + (float)inj[1];
      sr = t;
    }
  } else if (wave == 1) {                // backward: T_out[c], c = NC-1..0
    const float wr = pWQ[lane], wi = pWQ[N2 + lane];
    float sr = 0.f, si = 0.f;
    f16* base = Sb + ((lane >> 3) * NC) * 16 + ((2 * lane) & 15);
    for (int c = NC - 1; c >= 0; --c) {
      const f16x2 inj = *(const f16x2*)&base[c * 16];
      f16x2 out; out[0] = (f16)sr; out[1] = (f16)si;
      *(f16x2*)&base[c * 16] = out;
      const float t = wr * sr - wi * si + (float)inj[0];
      si = wi * sr + wr * si + (float)inj[1];
      sr = t;
    }
  }
  __syncthreads();

  // ---- output tiles: y(Q x NC) = T0@Uc + T1@Uc + Of@Sf + Ob@Sb ----
  // two independent accumulators -> WMMAs issue back-to-back (no D->C chain)
  for (int t = wave; t < 16; t += 4) {
    const int mt = t >> 3;               // which 16 rows within the chunk
    const int nt = t & 7;                // which 16 chunks
    v8f acc0 = {}, acc1 = {};
    acc0 = wmma_acc(pT0, Q,  16 * mt, 0,  Uc, 0, 16 * nt, acc0, g, m);
    acc1 = wmma_acc(pT1, Q,  16 * mt, 0,  Uc, 0, 16 * nt, acc1, g, m);
    acc0 = wmma_acc(pOF, 64, 16 * mt, 0,  Sf, 0, 16 * nt, acc0, g, m);
    acc1 = wmma_acc(pOF, 64, 16 * mt, 32, Sf, 2, 16 * nt, acc1, g, m);
    acc0 = wmma_acc(pOB, 64, 16 * mt, 0,  Sb, 0, 16 * nt, acc0, g, m);
    acc1 = wmma_acc(pOB, 64, 16 * mt, 32, Sb, 2, 16 * nt, acc1, g, m);
#pragma unroll
    for (int r = 0; r < 8; ++r)          // 8 consecutive f32 -> 2x ds_store_b128
      yS[(16 * nt + m) * Q + 16 * mt + 8 * g + r] = acc0[r] + acc1[r];
  }
  __syncthreads();

  // ---- y += D*u, coalesced float4 store ----
  const float Dh = Dv[h];
  float* yrow = y + ((size_t)b * HH + h) * LL;
  for (int i = tid; i < LL / 4; i += 128) {
    float4 uu = ((const float4*)uS)[i];
    float4 yy = ((const float4*)yS)[i];
    yy.x += Dh * uu.x; yy.y += Dh * uu.y;
    yy.z += Dh * uu.z; yy.w += Dh * uu.w;
    ((float4*)yrow)[i] = yy;
  }
}

// =====================================================================
extern "C" void kernel_launch(void* const* d_in, const int* in_sizes, int n_in,
                              void* d_out, int out_size, void* d_ws, size_t ws_size,
                              hipStream_t stream)
{
  const float* u   = (const float*)d_in[0];
  const float* ldt = (const float*)d_in[1];
  const float* lna = (const float*)d_in[2];
  const float* aim = (const float*)d_in[3];
  const float* Bp  = (const float*)d_in[4];
  const float* Cp  = (const float*)d_in[5];
  const float* Dv  = (const float*)d_in[6];
  float* yo = (float*)d_out;
  char*  ws = (char*)d_ws;   // needs 512 * 20736 B ~= 10.1 MB

  hipFuncSetAttribute((const void*)s4d_main,
                      hipFuncAttributeMaxDynamicSharedMemorySize, SMEM_MAIN);

  s4d_precompute<<<HH, 32, 0, stream>>>(ldt, lna, aim, Bp, Cp, ws);
  s4d_main<<<BB * HH, 128, SMEM_MAIN, stream>>>(u, Dv, ws, yo);
}
